// Bottleneck_26121991094509
// MI455X (gfx1250) — matrix operand order repaired and checked by reading, not compiled, not run on hardware
//
#include <hip/hip_runtime.h>
#include <hip/hip_bf16.h>

// ---------------- problem constants ----------------
static constexpr int NN   = 200000;  // sparse voxels
static constexpr int CIN  = 256;     // channels in/out
static constexpr int CMID = 64;      // bottleneck channels
static constexpr int KK   = 27;      // kernel offsets
static constexpr int MM   = 100000;  // pairs per offset
static constexpr int NB   = 4;       // batch instances
static constexpr float EPS = 1e-5f;
static constexpr int BPK  = (MM + 127) / 128;   // conv2 blocks per kernel offset = 782

// ---------------- WMMA types ----------------
typedef __attribute__((ext_vector_type(16))) __bf16 v16bf;
typedef __attribute__((ext_vector_type(8)))  float  v8f;

union ABu { v16bf v; uint4 q[2]; };

__device__ inline v8f wmma_bf16(v16bf a, v16bf b, v8f c) {
  // 8 args: (neg_a, A, neg_b, B, c_mod, C, reuse_a, reuse_b)
  return __builtin_amdgcn_wmma_f32_16x16x32_bf16(false, a, false, b, (short)0, c, false, false);
}

// round-to-nearest-even f32 -> bf16 bit pattern
__device__ inline unsigned short f2bf(float x) {
  unsigned int u = __float_as_uint(x);
  return (unsigned short)((u + 0x7FFFu + ((u >> 16) & 1u)) >> 16);
}

__device__ inline void atomAddF(float* p, float v) {
  __hip_atomic_fetch_add(p, v, __ATOMIC_RELAXED, __HIP_MEMORY_SCOPE_AGENT);
}

// ---------------- utility kernels ----------------
__global__ void k_zero(float* __restrict__ p, long n) {
  long stride = (long)gridDim.x * blockDim.x;
  for (long i = (long)blockIdx.x * blockDim.x + threadIdx.x; i < n; i += stride) p[i] = 0.f;
}

// Pre-swizzle weight matrix (nMat x Cin x Cout, row major f32) into WMMA B-operand
// layout, bf16. Flat index j: e = j&15 (element in v16bf), lane = (j>>4)&31,
// tile = j>>9 -> nt (Cout/16), kt (Cin/32), mat. Element e of lane L holds
// B[K = kt*32 + 8*(L>>4) + e + (e&8)][n = nt*16 + (L&15)]   (ISA 7.12.2 B layout).
__global__ void k_preswiz(const float* __restrict__ W, unsigned short* __restrict__ out,
                          int Cin, int Cout, int nMat) {
  int NT = Cout >> 4, KT = Cin >> 5;
  long total = (long)nMat * Cin * Cout;
  long stride = (long)gridDim.x * blockDim.x;
  for (long j = (long)blockIdx.x * blockDim.x + threadIdx.x; j < total; j += stride) {
    int e    = (int)(j & 15);
    int lane = (int)((j >> 4) & 31);
    long tile = j >> 9;
    int nt  = (int)(tile % NT);
    long t2 = tile / NT;
    int kt  = (int)(t2 % KT);
    int mat = (int)(t2 / KT);
    int n = lane & 15, h = lane >> 4;
    int Kk = kt * 32 + h * 8 + e + (e & 8);
    out[j] = f2bf(W[((long)mat * Cin + Kk) * Cout + nt * 16 + n]);
  }
}

// per-instance row counts (batch_idx is sorted -> flush trick)
__global__ void k_count(const int* __restrict__ bidx, float* __restrict__ cnt) {
  int stride = gridDim.x * blockDim.x;
  int cur = -1; float c = 0.f;
  for (int i = blockIdx.x * blockDim.x + threadIdx.x; i < NN; i += stride) {
    int b = bidx[i];
    if (b != cur) { if (cur >= 0) atomAddF(&cnt[cur], c); cur = b; c = 0.f; }
    c += 1.f;
  }
  if (cur >= 0) atomAddF(&cnt[cur], c);
}

// per-instance per-channel sum / sumsq over x[NN, C]; batch_idx sorted ->
// per-thread register accumulation, flush atomics only at instance boundaries.
__global__ void k_stats(const float* __restrict__ x, const int* __restrict__ bidx,
                        float* __restrict__ sum, float* __restrict__ sq,
                        int C, int rowsPerBlock) {
  int t = threadIdx.x;
  int tpc = blockDim.x / C;         // row-groups per block
  int c = t % C, rg = t / C;
  int base = blockIdx.x * rowsPerBlock;
  int rend = base + rowsPerBlock; if (rend > NN) rend = NN;
  int cur = -1; float s = 0.f, q = 0.f;
  for (int r = base + rg; r < rend; r += tpc) {
    int b = bidx[r];
    float v = x[(long)r * C + c];
    if (b != cur) {
      if (cur >= 0) { atomAddF(&sum[cur * C + c], s); atomAddF(&sq[cur * C + c], q); }
      cur = b; s = 0.f; q = 0.f;
    }
    s += v; q += v * v;
  }
  if (cur >= 0) { atomAddF(&sum[cur * C + c], s); atomAddF(&sq[cur * C + c], q); }
}

__global__ void k_finalize(const float* __restrict__ sum, const float* __restrict__ sq,
                           const float* __restrict__ cnt,
                           float* __restrict__ mean, float* __restrict__ scale, int C) {
  int t = blockIdx.x * blockDim.x + threadIdx.x;
  if (t < 4 * C) {
    int b = t / C;
    float inv = 1.0f / cnt[b];
    float m = sum[t] * inv;
    float var = sq[t] * inv - m * m;
    mean[t] = m;
    scale[t] = rsqrtf(fmaxf(var, 0.f) + EPS);
  }
}

// normalize + relu + convert h1 -> x2 (bf16)
__global__ void k_normrelu(const float* __restrict__ h1, const int* __restrict__ bidx,
                           const float* __restrict__ mean, const float* __restrict__ scale,
                           unsigned short* __restrict__ x2) {
  long n = (long)NN * CMID;
  long stride = (long)gridDim.x * blockDim.x;
  for (long i = (long)blockIdx.x * blockDim.x + threadIdx.x; i < n; i += stride) {
    int r = (int)(i >> 6), c = (int)(i & 63);
    int b = bidx[r];
    float v = (h1[i] - mean[b * 64 + c]) * scale[b * 64 + c];
    x2[i] = f2bf(fmaxf(v, 0.f));
  }
}

// out = relu((y - mean)*scale + feats), in place on y (= d_out)
__global__ void k_final(const float* __restrict__ feats, const int* __restrict__ bidx,
                        const float* __restrict__ mean, const float* __restrict__ scale,
                        float* __restrict__ y) {
  long n = (long)NN * CIN;
  long stride = (long)gridDim.x * blockDim.x;
  for (long i = (long)blockIdx.x * blockDim.x + threadIdx.x; i < n; i += stride) {
    int r = (int)(i >> 8), c = (int)(i & 255);
    int b = bidx[r];
    float v = (y[i] - mean[b * 256 + c]) * scale[b * 256 + c] + feats[i];
    y[i] = fmaxf(v, 0.f);
  }
}

// ---------------- GEMM1: h1[N,64] = feats[N,256] @ W1 (bf16 WMMA) ----------------
// block = 256 (8 waves), tile = 32 rows x 64 cols; wave (mt, nt) does 16x16 tile, K=256.
__global__ __launch_bounds__(256) void k_gemm1(const float* __restrict__ feats,
                                               const unsigned short* __restrict__ w1r,
                                               float* __restrict__ h1) {
  __shared__ alignas(16) unsigned short tA[32 * 264];   // 32 rows x 256 (+8 pad) bf16
  int t = threadIdx.x;
  int m0 = blockIdx.x * 32;
  #pragma unroll
  for (int i = 0; i < 32; ++i) {
    int idx = i * 256 + t;
    int r = idx >> 8, c = idx & 255;
    tA[r * 264 + c] = f2bf(feats[(long)(m0 + r) * 256 + c]);
  }
  __syncthreads();
  int w = t >> 5, lane = t & 31;
  int mt = w >> 2, nt = w & 3;
  int m = lane & 15, h = lane >> 4;
  v8f acc;
  #pragma unroll
  for (int i = 0; i < 8; ++i) acc[i] = 0.f;
  const unsigned short* arow = &tA[(mt * 16 + m) * 264 + 8 * h];
  #pragma unroll
  for (int kt = 0; kt < 8; ++kt) {
    ABu A;
    A.q[0] = *(const uint4*)(arow + kt * 32);        // K = kt*32 + 8h + 0..7
    A.q[1] = *(const uint4*)(arow + kt * 32 + 16);   // K = kt*32 + 16 + 8h + 0..7
    ABu Bv;
    const uint4* bp = (const uint4*)(w1r + ((kt * 4 + nt) * 512 + lane * 16));
    Bv.q[0] = bp[0]; Bv.q[1] = bp[1];
    acc = wmma_bf16(A.v, Bv.v, acc);
  }
  int n = lane & 15;
  int rbase = m0 + mt * 16 + 8 * h;   // D: VGPR r -> row = r + 8h, col = lane&15
  #pragma unroll
  for (int r = 0; r < 8; ++r) h1[(long)(rbase + r) * 64 + nt * 16 + n] = acc[r];
}

// ---------------- conv2: gather -> 64x64 bf16 WMMA -> scatter-add ----------------
// grid = KK * BPK blocks, block = 256 (8 waves), each wave owns one 16-row m-tile.
__global__ __launch_bounds__(256) void k_conv2(const unsigned short* __restrict__ x2,
                                               const unsigned short* __restrict__ w2r,
                                               const int* __restrict__ in_idx,
                                               const int* __restrict__ out_idx,
                                               float* __restrict__ h2) {
  __shared__ alignas(16) unsigned short tA[8][16 * 72];  // per-wave gathered tile
  int t = threadIdx.x, w = t >> 5, lane = t & 31;
  int k  = blockIdx.x / BPK;
  int mb = blockIdx.x % BPK;
  int m0 = mb * 128 + w * 16;
  if (m0 >= MM) return;                      // tail waves idle (M % 16 == 0)
  // gather: 2 lanes per row, 64B each, coalesced b128 copies to LDS
  int rr = lane >> 1, hh = lane & 1;
  int grow = in_idx[(long)k * MM + m0 + rr];
  const uint4* src = (const uint4*)(x2 + (long)grow * 64 + hh * 32);
  uint4* dst = (uint4*)(&tA[w][rr * 72 + hh * 32]);
  dst[0] = src[0]; dst[1] = src[1]; dst[2] = src[2]; dst[3] = src[3];
  __builtin_amdgcn_wave_barrier();           // per-wave; DS ops are in-order in HW
  int m = lane & 15, h = lane >> 4;
  v8f acc[4];
  #pragma unroll
  for (int nt = 0; nt < 4; ++nt)
    #pragma unroll
    for (int i = 0; i < 8; ++i) acc[nt][i] = 0.f;
  const unsigned short* arow = &tA[w][m * 72 + 8 * h];
  #pragma unroll
  for (int kt = 0; kt < 2; ++kt) {
    ABu A;
    A.q[0] = *(const uint4*)(arow + kt * 32);
    A.q[1] = *(const uint4*)(arow + kt * 32 + 16);
    #pragma unroll
    for (int nt = 0; nt < 4; ++nt) {
      ABu Bv;
      const uint4* bp = (const uint4*)(w2r + (((long)(k * 2 + kt) * 4 + nt) * 512 + lane * 16));
      Bv.q[0] = bp[0]; Bv.q[1] = bp[1];
      acc[nt] = wmma_bf16(A.v, Bv.v, acc[nt]);
    }
  }
  int n = lane & 15;
  #pragma unroll
  for (int r = 0; r < 8; ++r) {
    int orow = out_idx[(long)k * MM + m0 + 8 * h + r];
    float* dp = h2 + (long)orow * 64 + n;
    atomAddF(dp +  0, acc[0][r]);
    atomAddF(dp + 16, acc[1][r]);
    atomAddF(dp + 32, acc[2][r]);
    atomAddF(dp + 48, acc[3][r]);
  }
}

// ---------------- GEMM3: y[N,256] = relu(IN(h2)) @ W3, staged in d_out ----------------
// block = 256 (8 waves), 128 rows per block; normalize+relu+bf16 during staging.
__global__ __launch_bounds__(256) void k_gemm3(const float* __restrict__ h2,
                                               const int* __restrict__ bidx,
                                               const float* __restrict__ mean2,
                                               const float* __restrict__ scale2,
                                               const unsigned short* __restrict__ w3r,
                                               float* __restrict__ y) {
  __shared__ alignas(16) unsigned short tA[128 * 72];
  int t = threadIdx.x;
  int m0 = blockIdx.x * 128;
  int rows = NN - m0; if (rows > 128) rows = 128;
  #pragma unroll
  for (int i = 0; i < 32; ++i) {
    int idx = i * 256 + t;
    int r = idx >> 6, c = idx & 63;
    if (r < rows) {
      int b = bidx[m0 + r];
      float v = (h2[(long)(m0 + r) * 64 + c] - mean2[b * 64 + c]) * scale2[b * 64 + c];
      tA[r * 72 + c] = f2bf(fmaxf(v, 0.f));
    }
  }
  __syncthreads();
  int w = t >> 5, lane = t & 31;
  if (w * 16 >= rows) return;                 // tail (rows is a multiple of 16)
  int m = lane & 15, h = lane >> 4;
  const unsigned short* arow = &tA[(w * 16 + m) * 72 + 8 * h];
  ABu A0, A1;
  A0.q[0] = *(const uint4*)(arow);       A0.q[1] = *(const uint4*)(arow + 16);
  A1.q[0] = *(const uint4*)(arow + 32);  A1.q[1] = *(const uint4*)(arow + 48);
  int n = lane & 15;
  int rbase = m0 + w * 16 + 8 * h;
  #pragma unroll
  for (int nt = 0; nt < 16; ++nt) {
    v8f acc;
    #pragma unroll
    for (int i = 0; i < 8; ++i) acc[i] = 0.f;
    ABu B0, B1;
    const uint4* b0 = (const uint4*)(w3r + ((long)(0 * 16 + nt) * 512 + lane * 16));
    const uint4* b1 = (const uint4*)(w3r + ((long)(1 * 16 + nt) * 512 + lane * 16));
    B0.q[0] = b0[0]; B0.q[1] = b0[1];
    B1.q[0] = b1[0]; B1.q[1] = b1[1];
    acc = wmma_bf16(A0.v, B0.v, acc);
    acc = wmma_bf16(A1.v, B1.v, acc);
    #pragma unroll
    for (int r = 0; r < 8; ++r) y[(long)(rbase + r) * 256 + nt * 16 + n] = acc[r];
  }
}

// ---------------- host launcher ----------------
extern "C" void kernel_launch(void* const* d_in, const int* in_sizes, int n_in,
                              void* d_out, int out_size, void* d_ws, size_t ws_size,
                              hipStream_t stream) {
  (void)in_sizes; (void)n_in; (void)out_size; (void)ws_size;
  const float* feats  = (const float*)d_in[0];
  const float* W1     = (const float*)d_in[1];
  const float* W2     = (const float*)d_in[2];
  const float* W3     = (const float*)d_in[3];
  const int*   in_idx = (const int*)d_in[4];
  const int*   out_idx= (const int*)d_in[5];
  const int*   bidx   = (const int*)d_in[6];
  float* y = (float*)d_out;

  char* ws = (char*)d_ws;
  size_t off = 0;
  auto alloc = [&](size_t bytes) -> char* {
    char* p = ws + off; off += (bytes + 255) & ~(size_t)255; return p;
  };
  float*          h1  = (float*)          alloc((size_t)NN * 64 * 4);   // 51.2 MB
  unsigned short* x2  = (unsigned short*) alloc((size_t)NN * 64 * 2);   // 25.6 MB
  float*          h2  = (float*)          alloc((size_t)NN * 64 * 4);   // 51.2 MB
  unsigned short* w1r = (unsigned short*) alloc((size_t)256 * 64 * 2);
  unsigned short* w2r = (unsigned short*) alloc((size_t)KK * 64 * 64 * 2);
  unsigned short* w3r = (unsigned short*) alloc((size_t)64 * 256 * 2);
  float* stats = (float*)alloc(3076 * 4);  // sum1,sq1,sum2,sq2 (256 ea), sum3,sq3 (1024 ea), cnt(4)
  float* sum1 = stats;        float* sq1 = stats + 256;
  float* sum2 = stats + 512;  float* sq2 = stats + 768;
  float* sum3 = stats + 1024; float* sq3 = stats + 2048;
  float* cnt  = stats + 3072;
  float* mean1  = (float*)alloc(256 * 4);  float* scale1 = (float*)alloc(256 * 4);
  float* mean2  = (float*)alloc(256 * 4);  float* scale2 = (float*)alloc(256 * 4);
  float* mean3  = (float*)alloc(1024 * 4); float* scale3 = (float*)alloc(1024 * 4);

  // zero accumulators (deterministic across replays)
  k_zero<<<2048, 256, 0, stream>>>(h2, (long)NN * 64);
  k_zero<<<16,   256, 0, stream>>>(stats, 3076);

  // pre-swizzle weights into WMMA B layout
  k_preswiz<<<64,  256, 0, stream>>>(W1, w1r, 256, 64, 1);
  k_preswiz<<<432, 256, 0, stream>>>(W2, w2r, 64, 64, KK);
  k_preswiz<<<64,  256, 0, stream>>>(W3, w3r, 64, 256, 1);

  k_count<<<256, 256, 0, stream>>>(bidx, cnt);

  // conv1 -> IN -> ReLU
  k_gemm1<<<NN / 32, 256, 0, stream>>>(feats, w1r, h1);
  k_stats<<<(NN + 255) / 256, 256, 0, stream>>>(h1, bidx, sum1, sq1, 64, 256);
  k_finalize<<<1, 256, 0, stream>>>(sum1, sq1, cnt, mean1, scale1, 64);
  k_normrelu<<<4096, 256, 0, stream>>>(h1, bidx, mean1, scale1, x2);

  // sparse conv2 (gather-WMMA-scatter) -> IN -> ReLU (fused into gemm3 staging)
  k_conv2<<<KK * BPK, 256, 0, stream>>>(x2, w2r, in_idx, out_idx, h2);
  k_stats<<<(NN + 255) / 256, 256, 0, stream>>>(h2, bidx, sum2, sq2, 64, 256);
  k_finalize<<<1, 256, 0, stream>>>(sum2, sq2, cnt, mean2, scale2, 64);

  // conv3 -> IN -> residual -> ReLU (y staged in d_out, normalized in place)
  k_gemm3<<<(NN + 127) / 128, 256, 0, stream>>>(h2, bidx, mean2, scale2, w3r, y);
  k_stats<<<(NN + 63) / 64, 256, 0, stream>>>(y, bidx, sum3, sq3, 256, 64);
  k_finalize<<<1, 1024, 0, stream>>>(sum3, sq3, cnt, mean3, scale3, 256);
  k_final<<<4096, 256, 0, stream>>>(feats, bidx, mean3, scale3, y);
}
